// MyNNUE_6262062317932
// MI455X (gfx1250) — matrix operand order repaired and checked by reading, not compiled, not run on hardware
//
#include <hip/hip_runtime.h>

// ---------- types ----------
typedef __attribute__((ext_vector_type(16))) __bf16          v16bf;
typedef __attribute__((ext_vector_type(8)))  float           v8f;
typedef __attribute__((ext_vector_type(8)))  unsigned short  u16x8;
typedef __attribute__((ext_vector_type(4)))  unsigned short  u16x4;

struct BPair { u16x8 lo; u16x8 hi; };   // 32 bytes == v16bf
static_assert(sizeof(BPair) == sizeof(v16bf), "pair/v16bf size mismatch");

static __device__ __forceinline__ unsigned short f2bf(float f) {
  unsigned u = __builtin_bit_cast(unsigned, f);
  u += 0x7FFFu + ((u >> 16) & 1u);          // round-to-nearest-even
  return (unsigned short)(u >> 16);
}

#define COUNT   8
#define L1      512
#define L2N     16
#define L3N     32
#define ACCW    1040          // accum row width in f32
#define BROWS   128           // batch rows per workgroup
#define KC      128           // K chunk
#define NCHUNK  8             // 1024 / KC
#define PADK    136           // LDS halfword row stride (128 + 8 pad)

// ---------- LDS layout (bytes) ----------
#define OFF_SA    0                               // 128*136 u16  = 34816
#define OFF_SW    (OFF_SA + BROWS*PADK*2)         // 128*136 u16  = 34816
#define OFF_L1Y   (OFF_SW + 128*PADK*2)           // 128*17 f32   = 8704
#define OFF_L2W   (OFF_L1Y + BROWS*17*4)          // 256*16 f32   = 16384
#define OFF_L2B   (OFF_L2W + 256*16*4)            // 256 f32      = 1024
#define OFF_OUTW  (OFF_L2B + 256*4)               // 256 f32      = 1024
#define OFF_OUTB  (OFF_OUTW + 256*4)              // 16 f32       = 64
#define OFF_L1B   (OFF_OUTB + 16*4)               // 128 f32      = 512
#define OFF_LSI   (OFF_L1B + 128*4)               // 128 i32      = 512
#define SMEM_BYTES (OFF_LSI + 128*4)              // = 97856

// ---------- prologue: l1_w f32 -> bf16 ----------
__global__ void wcvt_kernel(const float* __restrict__ w,
                            unsigned short* __restrict__ o, int n) {
  int i = blockIdx.x * blockDim.x + threadIdx.x;
  if (i < n) o[i] = f2bf(w[i]);
}

// ---------- main fused kernel ----------
__global__ __launch_bounds__(256)
void nnue_kernel(const float* __restrict__ accum,
                 const float* __restrict__ us,
                 const float* __restrict__ l1_b,
                 const float* __restrict__ l2_w,
                 const float* __restrict__ l2_b,
                 const float* __restrict__ out_w,
                 const float* __restrict__ out_b,
                 const int*   __restrict__ psqt,
                 const int*   __restrict__ lsi,
                 const unsigned short* __restrict__ w1bf,
                 float* __restrict__ out) {
  extern __shared__ char smem[];
  unsigned short* sA   = (unsigned short*)(smem + OFF_SA);
  unsigned short* sW   = (unsigned short*)(smem + OFF_SW);
  float*          sL1y = (float*)(smem + OFF_L1Y);   // stride 17
  float*          sL2w = (float*)(smem + OFF_L2W);
  float*          sL2b = (float*)(smem + OFF_L2B);
  float*          sOutW= (float*)(smem + OFF_OUTW);
  float*          sOutB= (float*)(smem + OFF_OUTB);
  float*          sL1b = (float*)(smem + OFF_L1B);
  int*            sLsi = (int*)(smem + OFF_LSI);

  const int t    = threadIdx.x;
  const int wave = t >> 5;
  const int lane = t & 31;
  const int hi   = lane >> 4;        // half-wave select
  const int lm   = lane & 15;
  const long rowbase = (long)blockIdx.x * BROWS;

  // one-time small staging
  if (t < 128) { sLsi[t] = lsi[rowbase + t]; sL1b[t] = l1_b[t]; }
  for (int i = t; i < 256 * 16; i += 256) sL2w[i] = l2_w[i];
  if (t < 256) { sL2b[t] = l2_b[t]; sOutW[t] = out_w[t]; }
  if (t < COUNT) sOutB[t] = out_b[t];

  v8f c[COUNT];
  const v8f vzero = {0.f,0.f,0.f,0.f,0.f,0.f,0.f,0.f};
#pragma unroll
  for (int n = 0; n < COUNT; ++n) c[n] = vzero;

  const int srow16 = t >> 4;   // 0..15
  const int scol   = t & 15;   // 0..15

  for (int ch = 0; ch < NCHUNK; ++ch) {
    // ---- stage A: rows 0..127, K cols ch*128..+127, f32 -> bf16 ----
    const long gcol = (long)ch * KC + (ch >= 4 ? COUNT : 0); // skip psqt gap
#pragma unroll
    for (int it = 0; it < 8; ++it) {
      const int r = it * 16 + srow16;
      const float4* src = (const float4*)(accum + (rowbase + r) * ACCW + gcol);
      unsigned short* dst = sA + r * PADK;
#pragma unroll
      for (int h = 0; h < 2; ++h) {
        float4 v = src[h * 16 + scol];
        u16x4 p;
        p.x = f2bf(v.x); p.y = f2bf(v.y); p.z = f2bf(v.z); p.w = f2bf(v.w);
        *(u16x4*)(dst + h * 64 + scol * 4) = p;
      }
    }
    // ---- stage W slice (bf16 copy) ----
#pragma unroll
    for (int it = 0; it < 8; ++it) {
      const int r = it * 16 + srow16;
      u16x8 v = ((const u16x8*)(w1bf + r * 1024 + ch * KC))[scol];
      *(u16x8*)(sW + r * PADK + scol * 8) = v;
    }
    __syncthreads();

    // ---- GEMM: this wave's 16-row M tile vs all 8 N tiles ----
#pragma unroll
    for (int kt = 0; kt < 4; ++kt) {
      // A fragment: lane holds row M=lm, K = kt*32 + hi*8 + {0..7, 16..23}
      const unsigned short* ap = sA + (wave * 16 + lm) * PADK + kt * 32 + hi * 8;
      BPair apair;
      apair.lo = *(const u16x8*)(ap);
      apair.hi = *(const u16x8*)(ap + 16);
      v16bf av = __builtin_bit_cast(v16bf, apair);
#pragma unroll
      for (int n = 0; n < COUNT; ++n) {
        // B fragment: lane holds col N=lm of tile n, K = kt*32 + hi*8 + {0..7, 16..23}
        const unsigned short* bp = sW + (n * 16 + lm) * PADK + kt * 32 + hi * 8;
        BPair bpair;
        bpair.lo = *(const u16x8*)(bp);
        bpair.hi = *(const u16x8*)(bp + 16);
        v16bf bv = __builtin_bit_cast(v16bf, bpair);
        c[n] = __builtin_amdgcn_wmma_f32_16x16x32_bf16(
            false, av, false, bv, (short)0, c[n], false, false);
      }
    }
    __syncthreads();
  }

  // ---- epilogue: per-row tile selection + bias + clip -> sL1y ----
#pragma unroll
  for (int r = 0; r < 8; ++r) {
    const int m = wave * 16 + r + 8 * hi;  // local row (C layout: M=r+8*hi)
    const int sel = sLsi[m];
    float v = 0.f;
#pragma unroll
    for (int n = 0; n < COUNT; ++n) v = (sel == n) ? c[n][r] : v;
    v += sL1b[sel * 16 + lm];
    v = fminf(fmaxf(v, 0.f), 1.f);
    sL1y[m * 17 + lm] = v;
  }
  __syncthreads();

  // ---- tail: L2 (16->32), L3 (32->1), psqt ----
  if (t < BROWS) {
    const long row = rowbase + t;
    const int sl = sLsi[t];
    float y1[16];
#pragma unroll
    for (int k = 0; k < 16; ++k) y1[k] = sL1y[t * 17 + k];
    float acc = 0.f;
#pragma unroll
    for (int o = 0; o < L3N; ++o) {
      const int wr = sl * L3N + o;
      float s = sL2b[wr];
#pragma unroll
      for (int k = 0; k < 16; ++k) s = fmaf(y1[k], sL2w[wr * 16 + k], s);
      s = fminf(fmaxf(s, 0.f), 1.f);
      acc = fmaf(s, sOutW[wr], acc);
    }
    acc += sOutB[sl];
    const int pi = psqt[row];
    const float wps = accum[row * ACCW + L1 + pi];
    const float bps = accum[row * ACCW + 2 * L1 + COUNT + pi];
    out[row] = acc + (wps - bps) * (us[row] - 0.5f);
  }
}

// ---------- host entry ----------
extern "C" void kernel_launch(void* const* d_in, const int* in_sizes, int n_in,
                              void* d_out, int out_size, void* d_ws, size_t ws_size,
                              hipStream_t stream) {
  const float* accum = (const float*)d_in[0];
  const float* us    = (const float*)d_in[1];
  const float* l1_w  = (const float*)d_in[2];
  const float* l1_b  = (const float*)d_in[3];
  const float* l2_w  = (const float*)d_in[4];
  const float* l2_b  = (const float*)d_in[5];
  const float* out_w = (const float*)d_in[6];
  const float* out_b = (const float*)d_in[7];
  const int*   psqt  = (const int*)d_in[8];
  const int*   lsi   = (const int*)d_in[9];
  float* out = (float*)d_out;
  unsigned short* w1bf = (unsigned short*)d_ws;

  const int B = in_sizes[1];                 // us has B elements
  const int wn = in_sizes[2];                // 128*1024

  wcvt_kernel<<<(wn + 255) / 256, 256, 0, stream>>>(l1_w, w1bf, wn);

  const int blocks = B / BROWS;              // 131072/128 = 1024
  nnue_kernel<<<blocks, 256, SMEM_BYTES, stream>>>(
      accum, us, l1_b, l2_w, l2_b, out_w, out_b, psqt, lsi, w1bf, out);
}
